// DGCNN_83030307766414
// MI455X (gfx1250) — matrix operand order repaired and checked by reading, not compiled, not run on hardware
//
#include <hip/hip_runtime.h>
#include <float.h>

typedef __bf16 bf16_t;
typedef __attribute__((ext_vector_type(16))) __bf16 v16bf;
typedef __attribute__((ext_vector_type(8)))  __bf16 v8bf;
typedef __attribute__((ext_vector_type(8)))  float  v8f;

#define BATCH 8
#define NPTS  1024
#define KNBR  20
#define EPSBN 1e-5f

__device__ __forceinline__ float lrelu(float x) { return x >= 0.f ? x : 0.2f * x; }

// ---------------- KNN: per (b,n) block computes neg sq dist to all m, extracts top-20 ----
__global__ void k_knn(const float* __restrict__ X, long bstride, int C, int* __restrict__ idx)
{
    __shared__ float ctr[128];
    __shared__ float nd[NPTS];
    __shared__ float rv[128];
    __shared__ int   ri[128];
    int n = blockIdx.x, b = blockIdx.y, tid = threadIdx.x;
    const float* Xb = X + (long)b * bstride;
    for (int c = tid; c < C; c += 128) ctr[c] = Xb[(long)c * NPTS + n];
    __syncthreads();
    for (int m = tid; m < NPTS; m += 128) {
        float acc = 0.f;
        for (int c = 0; c < C; ++c) { float d = ctr[c] - Xb[(long)c * NPTS + m]; acc -= d * d; }
        nd[m] = acc;
    }
    __syncthreads();
    int* out = idx + ((long)b * NPTS + n) * KNBR;
    for (int it = 0; it < KNBR; ++it) {
        float best = -FLT_MAX; int bi = 0x7fffffff;
        for (int m = tid; m < NPTS; m += 128) {
            float v = nd[m];
            if (v > best || (v == best && m < bi)) { best = v; bi = m; }
        }
        rv[tid] = best; ri[tid] = bi;
        __syncthreads();
        for (int s = 64; s > 0; s >>= 1) {
            if (tid < s) {
                if (rv[tid + s] > rv[tid] || (rv[tid + s] == rv[tid] && ri[tid + s] < ri[tid])) {
                    rv[tid] = rv[tid + s]; ri[tid] = ri[tid + s];
                }
            }
            __syncthreads();
        }
        if (tid == 0) { out[it] = ri[0]; nd[ri[0]] = -FLT_MAX; }
        __syncthreads();
    }
}

// ---------------- bf16 WMMA GEMM: P[b] (MxN) = Ws (MxKd, row major) @ Xb[b] (rows of Kd per point) ----
__global__ void k_gemm_wmma(const bf16_t* __restrict__ Ws, const bf16_t* __restrict__ Xb,
                            float* __restrict__ P, int M, int Kd, int ldb, long bsX)
{
    int lane = threadIdx.x;                 // one wave32 per block
    int m0 = blockIdx.x << 4, n0 = blockIdx.y << 4, b = blockIdx.z;
    int row = lane & 15, half = lane >> 4;
    const bf16_t* Wr = Ws + (long)(m0 + row) * Kd;                       // A: lane holds row m0+row
    const bf16_t* Xc = Xb + (long)b * bsX + (long)(n0 + row) * ldb + half * 16; // B: lane holds col n0+row
    v8f acc = {0.f, 0.f, 0.f, 0.f, 0.f, 0.f, 0.f, 0.f};
    for (int kk = 0; kk < Kd; kk += 32) {
        // A fragment (16x32 bf16): elem i -> K = kk + 16*half + i
        v16bf a = *(const v16bf*)(Wr + kk + half * 16);
        // B fragment (32x16 bf16): elem i -> K = kk + 16*half + i (contiguous in transposed layout)
        v16bf bb = *(const v16bf*)(Xc + kk);
        acc = __builtin_amdgcn_wmma_f32_16x16x32_bf16(false, a, false, bb, (short)0, acc, false, false);
    }
    // D layout: m = m0 + 8*half + r, n = n0 + (lane&15)
    float* Pr = P + ((long)b * M + (m0 + half * 8)) * (long)NPTS + (n0 + row);
#pragma unroll
    for (int r = 0; r < 8; ++r) Pr[(long)r * NPTS] = acc[r];
}

// ---------------- BN stats over (b,n,k) of y = A[idx] + Bm, one block per channel (no atomics) ----
__global__ void k_edge_stats(const float* __restrict__ P, const int* __restrict__ idx,
                             int O, int M, float* __restrict__ stats)
{
    int o = blockIdx.x, tid = threadIdx.x;
    float s = 0.f, s2 = 0.f;
    for (int b = 0; b < BATCH; ++b) {
        const float* Ar = P + ((long)b * M + o) * NPTS;
        const float* Br = Ar + (long)O * NPTS;
        const int* ib = idx + (long)b * NPTS * KNBR;
        for (int j = tid; j < NPTS * KNBR; j += 256) {
            float v = Ar[ib[j]] + Br[j / KNBR];
            s += v; s2 += v * v;
        }
    }
    __shared__ float r1[256], r2[256];
    r1[tid] = s; r2[tid] = s2; __syncthreads();
    for (int st = 128; st > 0; st >>= 1) {
        if (tid < st) { r1[tid] += r1[tid + st]; r2[tid] += r2[tid + st]; }
        __syncthreads();
    }
    if (tid == 0) { stats[2 * o] = r1[0]; stats[2 * o + 1] = r2[0]; }
}

// ---------------- finalize: max_k (or min_k if g<0), BN affine, lrelu; write f32 hcat + bf16 hcatT ----
__global__ void k_edge_final(const float* __restrict__ P, const int* __restrict__ idx,
                             const float* __restrict__ stats, const float* __restrict__ g,
                             const float* __restrict__ bias, int O, int M, int coff,
                             float* __restrict__ hcat, bf16_t* __restrict__ hcatT)
{
    int o = blockIdx.x, b = blockIdx.y, tid = threadIdx.x;
    const float cnt = (float)(BATCH * NPTS * KNBR);
    float mean = stats[2 * o] / cnt;
    float var = stats[2 * o + 1] / cnt - mean * mean;
    float inv = rsqrtf(var + EPSBN);
    float gg = g[o];
    float scale = gg * inv;
    float shift = bias[o] - mean * scale;
    const float* Ar = P + ((long)b * M + o) * NPTS;
    const float* Br = Ar + (long)O * NPTS;
    const int* ib = idx + (long)b * NPTS * KNBR;
    float* ho = hcat + ((long)b * 512 + coff + o) * NPTS;
    bf16_t* ht = hcatT + (long)b * NPTS * 512 + coff + o;
    for (int n = tid; n < NPTS; n += 256) {
        const int* ii = ib + n * KNBR;
        float e;
        if (gg >= 0.f) { e = -FLT_MAX; for (int k = 0; k < KNBR; ++k) e = fmaxf(e, Ar[ii[k]]); }
        else           { e =  FLT_MAX; for (int k = 0; k < KNBR; ++k) e = fminf(e, Ar[ii[k]]); }
        float v = lrelu(scale * (e + Br[n]) + shift);
        ho[n] = v;
        ht[(long)n * 512] = (bf16_t)v;
    }
}

// ---------------- conv5 BN stats over (b,n), one block per channel ----
__global__ void k_stats5(const float* __restrict__ P5, float* __restrict__ stats5)
{
    int o = blockIdx.x, tid = threadIdx.x;
    float s = 0.f, s2 = 0.f;
    for (int b = 0; b < BATCH; ++b) {
        const float* r = P5 + ((long)b * 1024 + o) * NPTS;
        for (int n = tid; n < NPTS; n += 256) { float v = r[n]; s += v; s2 += v * v; }
    }
    __shared__ float r1[256], r2[256];
    r1[tid] = s; r2[tid] = s2; __syncthreads();
    for (int st = 128; st > 0; st >>= 1) {
        if (tid < st) { r1[tid] += r1[tid + st]; r2[tid] += r2[tid + st]; }
        __syncthreads();
    }
    if (tid == 0) { stats5[2 * o] = r1[0]; stats5[2 * o + 1] = r2[0]; }
}

__global__ void k_affine5(const float* __restrict__ stats5, const float* __restrict__ g,
                          const float* __restrict__ bias, float* __restrict__ scale,
                          float* __restrict__ shift)
{
    int o = blockIdx.x * blockDim.x + threadIdx.x;
    if (o >= 1024) return;
    const float cnt = (float)(BATCH * NPTS);
    float mean = stats5[2 * o] / cnt;
    float var = stats5[2 * o + 1] / cnt - mean * mean;
    float sc = g[o] * rsqrtf(var + EPSBN);
    scale[o] = sc;
    shift[o] = bias[o] - mean * sc;
}

// ---------------- global pooling: z = [max_n lrelu(...), mean_n lrelu(...)] ----
__global__ void k_zpool(const float* __restrict__ P5, const float* __restrict__ scale,
                        const float* __restrict__ shift, float* __restrict__ z)
{
    int o = blockIdx.x, b = blockIdx.y, tid = threadIdx.x;
    const float* r = P5 + ((long)b * 1024 + o) * NPTS;
    float sc = scale[o], sh = shift[o];
    float mx = -FLT_MAX, sm = 0.f;
    for (int n = tid; n < NPTS; n += 256) {
        float v = lrelu(sc * r[n] + sh);
        mx = fmaxf(mx, v); sm += v;
    }
    __shared__ float rm[256], rs[256];
    rm[tid] = mx; rs[tid] = sm; __syncthreads();
    for (int st = 128; st > 0; st >>= 1) {
        if (tid < st) { rm[tid] = fmaxf(rm[tid], rm[tid + st]); rs[tid] += rs[tid + st]; }
        __syncthreads();
    }
    if (tid == 0) {
        z[(long)b * 2048 + o] = rm[0];
        z[(long)b * 2048 + 1024 + o] = rs[0] * (1.f / NPTS);
    }
}

// ---------------- tiny FC head ----
__global__ void k_fc(const float* __restrict__ in, const float* __restrict__ W,
                     const float* __restrict__ bias, float* __restrict__ out, int Fout, int Fin)
{
    int t = blockIdx.x * blockDim.x + threadIdx.x;
    if (t >= Fout * BATCH) return;
    int j = t >> 3, s = t & 7;
    const float* ir = in + (long)s * Fin;
    const float* wr = W + (long)j * Fin;
    float acc = bias ? bias[j] : 0.f;
    for (int c = 0; c < Fin; ++c) acc += ir[c] * wr[c];
    out[(long)s * Fout + j] = acc;
}

__global__ void k_bn_head(const float* __restrict__ y, const float* __restrict__ g,
                          const float* __restrict__ bias, float* __restrict__ out, int F)
{
    int j = blockIdx.x * blockDim.x + threadIdx.x;
    if (j >= F) return;
    float s = 0.f, s2 = 0.f;
    for (int t = 0; t < BATCH; ++t) { float v = y[t * F + j]; s += v; s2 += v * v; }
    float mean = s / BATCH, var = s2 / BATCH - mean * mean;
    float sc = g[j] * rsqrtf(var + EPSBN);
    float sh = bias[j] - mean * sc;
    for (int t = 0; t < BATCH; ++t) out[t * F + j] = lrelu(sc * y[t * F + j] + sh);
}

// ---------------- weight / input prep ----
__global__ void k_stackW(const float* __restrict__ W, bf16_t* __restrict__ Ws, int O, int C, int Kp)
{
    int t = blockIdx.x * blockDim.x + threadIdx.x;
    if (t >= 2 * O * Kp) return;
    int m = t / Kp, kp = t - m * Kp;
    float v = 0.f;
    if (kp < C) v = (m < O) ? W[m * 2 * C + kp]
                            : (W[(m - O) * 2 * C + C + kp] - W[(m - O) * 2 * C + kp]);
    Ws[t] = (bf16_t)v;
}

__global__ void k_cvt(const float* __restrict__ in, bf16_t* __restrict__ out, int n)
{
    int t = blockIdx.x * blockDim.x + threadIdx.x;
    if (t < n) out[t] = (bf16_t)in[t];
}

__global__ void k_xb0(const float* __restrict__ x, bf16_t* __restrict__ xb)
{
    int t = blockIdx.x * blockDim.x + threadIdx.x;
    if (t >= BATCH * NPTS * 32) return;
    int b = t / (NPTS * 32); int r = t - b * NPTS * 32; int n = r >> 5; int c = r & 31;
    float v = (c < 3) ? x[((long)b * 3 + c) * NPTS + n] : 0.f;
    xb[t] = (bf16_t)v;
}

extern "C" void kernel_launch(void* const* d_in, const int* in_sizes, int n_in,
                              void* d_out, int out_size, void* d_ws, size_t ws_size,
                              hipStream_t stream)
{
    (void)in_sizes; (void)n_in; (void)out_size; (void)ws_size;
    const float* x   = (const float*)d_in[0];
    const float* W1  = (const float*)d_in[1];
    const float* W2  = (const float*)d_in[2];
    const float* W3  = (const float*)d_in[3];
    const float* W4  = (const float*)d_in[4];
    const float* W5  = (const float*)d_in[5];
    const float* Wl1 = (const float*)d_in[6];
    const float* Wl2 = (const float*)d_in[7];
    const float* bl2 = (const float*)d_in[8];
    const float* Wl3 = (const float*)d_in[9];
    const float* bl3 = (const float*)d_in[10];
    const float* g1 = (const float*)d_in[11]; const float* b1 = (const float*)d_in[12];
    const float* g2 = (const float*)d_in[13]; const float* b2 = (const float*)d_in[14];
    const float* g3 = (const float*)d_in[15]; const float* b3 = (const float*)d_in[16];
    const float* g4 = (const float*)d_in[17]; const float* b4 = (const float*)d_in[18];
    const float* g5 = (const float*)d_in[19]; const float* b5 = (const float*)d_in[20];
    const float* g6 = (const float*)d_in[21]; const float* b6 = (const float*)d_in[22];
    const float* g7 = (const float*)d_in[23]; const float* b7 = (const float*)d_in[24];

    char* ws = (char*)d_ws;
    float*  P     = (float*)(ws + 0);           // 32MB: GEMM out, reused by all layers + conv5
    float*  hcat  = (float*)(ws + 33554432);    // [8][512][1024] f32
    bf16_t* hcatT = (bf16_t*)(ws + 50331648);   // [8][1024][512] bf16
    bf16_t* xb0   = (bf16_t*)(ws + 58720256);   // [8][1024][32] bf16 (zero-padded 3->32)
    int*    idx   = (int*)(ws + 59244544);      // [8][1024][20], reused per layer
    bf16_t* Ws1   = (bf16_t*)(ws + 59899904);   // [128][32]
    bf16_t* Ws2   = (bf16_t*)(ws + 59908096);   // [128][64]
    bf16_t* Ws3   = (bf16_t*)(ws + 59924480);   // [256][64]
    bf16_t* Ws4   = (bf16_t*)(ws + 59957248);   // [512][128]
    bf16_t* W5b   = (bf16_t*)(ws + 60088320);   // [1024][512]
    float*  statsE= (float*)(ws + 61136896);    // up to 2*512 f32, reused per layer
    float*  stats5= (float*)(ws + 61140992);    // 2*1024 f32
    float*  sc5   = (float*)(ws + 61149184);    // 1024 f32
    float*  sh5   = sc5 + 1024;
    float*  z     = (float*)(ws + 61157376);    // [8][2048]
    float*  y1    = (float*)(ws + 61222912);    // [8][512]
    float*  a1    = (float*)(ws + 61239296);
    float*  y2    = (float*)(ws + 61255680);    // [8][256]
    float*  a2    = (float*)(ws + 61263872);
    float*  out   = (float*)d_out;              // [8][40]

    // ---- prep ----
    k_stackW<<<(128 * 32 + 255) / 256, 256, 0, stream>>>(W1, Ws1, 64, 3, 32);
    k_stackW<<<(128 * 64 + 255) / 256, 256, 0, stream>>>(W2, Ws2, 64, 64, 64);
    k_stackW<<<(256 * 64 + 255) / 256, 256, 0, stream>>>(W3, Ws3, 128, 64, 64);
    k_stackW<<<(512 * 128 + 255) / 256, 256, 0, stream>>>(W4, Ws4, 256, 128, 128);
    k_cvt<<<(1024 * 512 + 255) / 256, 256, 0, stream>>>(W5, W5b, 1024 * 512);
    k_xb0<<<(BATCH * NPTS * 32 + 255) / 256, 256, 0, stream>>>(x, xb0);

    // ---- layer 1: C=3(pad32), O=64 ----
    k_knn<<<dim3(NPTS, BATCH), 128, 0, stream>>>(x, (long)3 * NPTS, 3, idx);
    k_gemm_wmma<<<dim3(8, 64, BATCH), 32, 0, stream>>>(Ws1, xb0, P, 128, 32, 32, (long)NPTS * 32);
    k_edge_stats<<<64, 256, 0, stream>>>(P, idx, 64, 128, statsE);
    k_edge_final<<<dim3(64, BATCH), 256, 0, stream>>>(P, idx, statsE, g1, b1, 64, 128, 0, hcat, hcatT);

    // ---- layer 2: C=64, O=64 ----
    k_knn<<<dim3(NPTS, BATCH), 128, 0, stream>>>(hcat, (long)512 * NPTS, 64, idx);
    k_gemm_wmma<<<dim3(8, 64, BATCH), 32, 0, stream>>>(Ws2, hcatT, P, 128, 64, 512, (long)NPTS * 512);
    k_edge_stats<<<64, 256, 0, stream>>>(P, idx, 64, 128, statsE);
    k_edge_final<<<dim3(64, BATCH), 256, 0, stream>>>(P, idx, statsE, g2, b2, 64, 128, 64, hcat, hcatT);

    // ---- layer 3: C=64, O=128 ----
    k_knn<<<dim3(NPTS, BATCH), 128, 0, stream>>>(hcat + 64 * NPTS, (long)512 * NPTS, 64, idx);
    k_gemm_wmma<<<dim3(16, 64, BATCH), 32, 0, stream>>>(Ws3, hcatT + 64, P, 256, 64, 512, (long)NPTS * 512);
    k_edge_stats<<<128, 256, 0, stream>>>(P, idx, 128, 256, statsE);
    k_edge_final<<<dim3(128, BATCH), 256, 0, stream>>>(P, idx, statsE, g3, b3, 128, 256, 128, hcat, hcatT);

    // ---- layer 4: C=128, O=256 ----
    k_knn<<<dim3(NPTS, BATCH), 128, 0, stream>>>(hcat + 128 * NPTS, (long)512 * NPTS, 128, idx);
    k_gemm_wmma<<<dim3(32, 64, BATCH), 32, 0, stream>>>(Ws4, hcatT + 128, P, 512, 128, 512, (long)NPTS * 512);
    k_edge_stats<<<256, 256, 0, stream>>>(P, idx, 256, 512, statsE);
    k_edge_final<<<dim3(256, BATCH), 256, 0, stream>>>(P, idx, statsE, g4, b4, 256, 512, 256, hcat, hcatT);

    // ---- conv5: 512 -> 1024, BN over (b,n), lrelu, then max+mean pool ----
    k_gemm_wmma<<<dim3(64, 64, BATCH), 32, 0, stream>>>(W5b, hcatT, P, 1024, 512, 512, (long)NPTS * 512);
    k_stats5<<<1024, 256, 0, stream>>>(P, stats5);
    k_affine5<<<4, 256, 0, stream>>>(stats5, g5, b5, sc5, sh5);
    k_zpool<<<dim3(1024, BATCH), 256, 0, stream>>>(P, sc5, sh5, z);

    // ---- MLP head ----
    k_fc<<<(512 * BATCH + 255) / 256, 256, 0, stream>>>(z, Wl1, nullptr, y1, 512, 2048);
    k_bn_head<<<2, 256, 0, stream>>>(y1, g6, b6, a1, 512);
    k_fc<<<(256 * BATCH + 255) / 256, 256, 0, stream>>>(a1, Wl2, bl2, y2, 256, 512);
    k_bn_head<<<1, 256, 0, stream>>>(y2, g7, b7, a2, 256);
    k_fc<<<(40 * BATCH + 255) / 256, 256, 0, stream>>>(a2, Wl3, bl3, out, 40, 256);
}
